// RNN_rand_7499012899289
// MI455X (gfx1250) — matrix operand order repaired and checked by reading, not compiled, not run on hardware
//
#include <hip/hip_runtime.h>

// ---------------------------------------------------------------------------
// 2-layer GRU (T=16384, H=384) + linear head for MI455X (gfx1250, wave32).
//
//   1. cvt:   y_meas, w_ih_0, w_ih_1  -> bf16 (RNE)
//   2. trans: w_hh_{0,1} [1152x384]   -> k-major bf16 [384x1152]
//   3. WMMA GEMM: ig = A_bf16 @ w_ih^T + b  -> f32 [T x 1152]
//        one wave per 16x64 strip, 4x v_wmma_f32_16x16x32_bf16 per k-step,
//        A fragment reused across the 4 N-tiles.
//   4. persistent scan: 1 WG x 384 threads. k-split x column-block matvec:
//        48 column-groups (8 cols) x 8 k-slices (48 k), b128 weight loads,
//        LDS partial reduction; ig gate inputs double-buffered via
//        global_load_async_to_lds_b32 + s_wait_asynccnt (ASYNCcnt).
//   5. final 64x384 matvec -> d_out (f32)
// ---------------------------------------------------------------------------

#define T_STEPS 16384
#define H       384
#define H3      1152
#define N_EFF   64

// scan decomposition: 48 groups of 8 columns x 8 k-slices of 48
#define GCOL 8
#define NGRP 48
#define NSLC 8
#define KSL  48

typedef __attribute__((ext_vector_type(16))) __bf16 v16bf;
typedef __attribute__((ext_vector_type(8)))  float  v8f;

// ---- manual bf16 helpers (storage = unsigned short, RNE rounding) ----------
__device__ __forceinline__ unsigned short f32_to_bf16(float f) {
    unsigned u = __float_as_uint(f);
    unsigned rnd = 0x7FFFu + ((u >> 16) & 1u);
    return (unsigned short)((u + rnd) >> 16);
}
__device__ __forceinline__ float bf16_lo(unsigned u) {
    return __uint_as_float(u << 16);
}
__device__ __forceinline__ float bf16_hi(unsigned u) {
    return __uint_as_float(u & 0xFFFF0000u);
}

// ---- elementwise f32 -> bf16 ----------------------------------------------
__global__ void cvt_f32_bf16(const float* __restrict__ src,
                             unsigned short* __restrict__ dst, int n) {
    int i = blockIdx.x * blockDim.x + threadIdx.x;
    if (i < n) dst[i] = f32_to_bf16(src[i]);
}

// ---- w_hh [1152 x 384] row-major f32 -> wT [384 x 1152] bf16 (k-major) ----
__global__ void whh_transpose_bf16(const float* __restrict__ w,
                                   unsigned short* __restrict__ wT) {
    int idx = blockIdx.x * blockDim.x + threadIdx.x;   // over 1152*384
    if (idx >= H3 * H) return;
    int k = idx / H3;
    int r = idx - k * H3;
    wT[idx] = f32_to_bf16(w[r * H + k]);
}

// ---- WMMA GEMM: C[M x 1152] = A[M x 384](bf16) @ W[1152 x 384]^T + bias ---
// One wave per 16(M) x 64(N) strip: A fragment reused by 4 WMMAs.
__global__ __launch_bounds__(32)
void wmma_gemm_bias(const unsigned short* __restrict__ A,
                    const unsigned short* __restrict__ W,
                    const float* __restrict__ bias,
                    float* __restrict__ C) {
    const int lane = threadIdx.x;          // 0..31
    const int r0 = blockIdx.x * 16;        // M tile
    const int n0 = blockIdx.y * 64;        // N strip (4 x 16)

    const int am      = lane & 15;
    const int a_khalf = (lane >> 4) ? 16 : 0;
    const int bn      = lane & 15;
    const int b_khalf = (lane >> 4) ? 16 : 0;

    const unsigned short* arow = A + (size_t)(r0 + am) * H;
    const unsigned short* brow = W + (size_t)(n0 + bn) * H;

    v8f acc0 = {}, acc1 = {}, acc2 = {}, acc3 = {};
#pragma unroll 2
    for (int k0 = 0; k0 < H; k0 += 32) {
        union { v16bf v; unsigned u[8]; } af, bf0, bf1, bf2, bf3;
#pragma unroll
        for (int v = 0; v < 8; ++v) {
            int ka = k0 + a_khalf + 2 * v;
            af.u[v] = *(const unsigned*)(arow + ka);
            int kb = k0 + b_khalf + 2 * v;
            bf0.u[v] = *(const unsigned*)(brow + kb);
            bf1.u[v] = *(const unsigned*)(brow + 16 * H + kb);
            bf2.u[v] = *(const unsigned*)(brow + 32 * H + kb);
            bf3.u[v] = *(const unsigned*)(brow + 48 * H + kb);
        }
        acc0 = __builtin_amdgcn_wmma_f32_16x16x32_bf16(false, af.v, false, bf0.v,
                                                       (short)0, acc0, false, false);
        acc1 = __builtin_amdgcn_wmma_f32_16x16x32_bf16(false, af.v, false, bf1.v,
                                                       (short)0, acc1, false, false);
        acc2 = __builtin_amdgcn_wmma_f32_16x16x32_bf16(false, af.v, false, bf2.v,
                                                       (short)0, acc2, false, false);
        acc3 = __builtin_amdgcn_wmma_f32_16x16x32_bf16(false, af.v, false, bf3.v,
                                                       (short)0, acc3, false, false);
    }

    const int cn    = lane & 15;
    const int mbase = (lane >> 4) ? 8 : 0;
    const v8f* accs[4] = { &acc0, &acc1, &acc2, &acc3 };
#pragma unroll
    for (int j = 0; j < 4; ++j) {
        const float bv = bias[n0 + j * 16 + cn];
        float* cp = C + (size_t)(r0 + mbase) * H3 + n0 + j * 16 + cn;
#pragma unroll
        for (int i = 0; i < 8; ++i)
            cp[(size_t)i * H3] = (*accs[j])[i] + bv;
    }
}

// ---- async ig prefetch: 3 b32 global->LDS copies for this thread ----------
__device__ __forceinline__ void async_ig_prefetch(const float* __restrict__ igp,
                                                  unsigned lds_base, int t) {
    unsigned l0 = lds_base + 4u * (unsigned)t;
    unsigned long long g0 = (unsigned long long)(uintptr_t)(igp + t);
    asm volatile("global_load_async_to_lds_b32 %0, %1, off"
                 :: "v"(l0), "v"(g0) : "memory");
    unsigned l1 = l0 + 4u * H;
    unsigned long long g1 = g0 + 4u * H;
    asm volatile("global_load_async_to_lds_b32 %0, %1, off"
                 :: "v"(l1), "v"(g1) : "memory");
    unsigned l2 = l0 + 8u * H;
    unsigned long long g2 = g0 + 8u * H;
    asm volatile("global_load_async_to_lds_b32 %0, %1, off"
                 :: "v"(l2), "v"(g2) : "memory");
}

// ---- persistent GRU scan: 1 workgroup, 384 threads (12 wave32) -------------
// Phase 1: thread (g = t%48, s = t/48) accumulates k in [48s, 48s+48) for
//          columns [8g, 8g+8) of all 3 gates, b128 weight loads, partials
//          into LDS red[s][gate][col].
// Phase 2: thread t owns column t: sums 8 slices/gate, gate math, h update.
__global__ __launch_bounds__(H)
void gru_scan(const float* __restrict__ ig,          // [T x 1152]
              const unsigned short* __restrict__ wT, // [384 x 1152] bf16
              const float* __restrict__ bn_vec,      // [384]
              unsigned short* __restrict__ seq_bf,   // [T x 384] bf16 or nullptr
              float* __restrict__ h_final) {         // [384]
    __shared__ float h[H];
    __shared__ float igbuf[2][H3];
    __shared__ float red[NSLC][3][H];

    const int t = threadIdx.x;
    const int g = t % NGRP;           // column group
    const int s = t / NGRP;           // k slice
    const int c0 = g * GCOL;
    const int k0 = s * KSL;

    h[t] = 0.0f;
    const float bnv = bn_vec[t];
    const unsigned buf_base0 = (unsigned)(uintptr_t)(void*)&igbuf[0][0];
    const unsigned buf_base1 = (unsigned)(uintptr_t)(void*)&igbuf[1][0];

    // kick off step 0 prefetch (completion guaranteed by in-loop waits)
    async_ig_prefetch(ig, buf_base0, t);
    __syncthreads();

    for (int step = 0; step < T_STEPS; ++step) {
        // issue next step's gate-input prefetch up front
        if (step + 1 < T_STEPS)
            async_ig_prefetch(ig + (size_t)(step + 1) * H3,
                              ((step + 1) & 1) ? buf_base1 : buf_base0, t);

        // ---- phase 1: partial matvec over this thread's k slice ----
        float2 acc[3][4];
#pragma unroll
        for (int gt = 0; gt < 3; ++gt)
#pragma unroll
            for (int p = 0; p < 4; ++p) acc[gt][p] = make_float2(0.0f, 0.0f);

        const unsigned short* wbase = wT + (size_t)k0 * H3 + c0;
#pragma unroll 2
        for (int kk = 0; kk < KSL; ++kk) {
            const float hk = h[k0 + kk];                 // LDS broadcast
            const unsigned short* wr = wbase + (size_t)kk * H3;
#pragma unroll
            for (int gt = 0; gt < 3; ++gt) {
                const uint4 wv = *(const uint4*)(wr + gt * H);  // 8 bf16, 16B
                acc[gt][0].x = fmaf(bf16_lo(wv.x), hk, acc[gt][0].x);
                acc[gt][0].y = fmaf(bf16_hi(wv.x), hk, acc[gt][0].y);
                acc[gt][1].x = fmaf(bf16_lo(wv.y), hk, acc[gt][1].x);
                acc[gt][1].y = fmaf(bf16_hi(wv.y), hk, acc[gt][1].y);
                acc[gt][2].x = fmaf(bf16_lo(wv.z), hk, acc[gt][2].x);
                acc[gt][2].y = fmaf(bf16_hi(wv.z), hk, acc[gt][2].y);
                acc[gt][3].x = fmaf(bf16_lo(wv.w), hk, acc[gt][3].x);
                acc[gt][3].y = fmaf(bf16_hi(wv.w), hk, acc[gt][3].y);
            }
        }
#pragma unroll
        for (int gt = 0; gt < 3; ++gt) {
            float4* rp = (float4*)&red[s][gt][c0];       // 16B-aligned (c0 % 8 == 0)
            rp[0] = make_float4(acc[gt][0].x, acc[gt][0].y, acc[gt][1].x, acc[gt][1].y);
            rp[1] = make_float4(acc[gt][2].x, acc[gt][2].y, acc[gt][3].x, acc[gt][3].y);
        }
        __syncthreads();   // partials visible

        // ---- wait for this step's async ig buffer (in-order completion) ----
        if (step + 1 < T_STEPS)
            asm volatile("s_wait_asynccnt 0x3" ::: "memory");
        else
            asm volatile("s_wait_asynccnt 0x0" ::: "memory");

        // ---- phase 2: thread t owns column t ----
        float hr = 0.0f, hz = 0.0f, hn = 0.0f;
#pragma unroll
        for (int sl = 0; sl < NSLC; ++sl) {
            hr += red[sl][0][t];
            hz += red[sl][1][t];
            hn += red[sl][2][t];
        }
        const float* igp = igbuf[step & 1];
        const float r = 1.0f / (1.0f + __expf(-(igp[t] + hr)));
        const float z = 1.0f / (1.0f + __expf(-(igp[t + H] + hz)));
        const float n = tanhf(igp[t + 2 * H] + r * (hn + bnv));
        const float hold = h[t];
        const float hnew = n + z * (hold - n);
        __syncthreads();   // everyone done reading h and red
        h[t] = hnew;
        if (seq_bf) seq_bf[(size_t)step * H + t] = f32_to_bf16(hnew);
        __syncthreads();   // h fully updated for next step
    }
    h_final[t] = h[t];
}

// ---- out[j] = lin_w[j,:] . h_final + lin_b[j] ------------------------------
__global__ __launch_bounds__(N_EFF)
void final_linear(const float* __restrict__ hf, const float* __restrict__ lw,
                  const float* __restrict__ lb, float* __restrict__ out) {
    int j = threadIdx.x;
    if (j >= N_EFF) return;
    float s = lb[j];
#pragma unroll 4
    for (int k = 0; k < H; ++k) s = fmaf(lw[j * H + k], hf[k], s);
    out[j] = s;
}

// ---------------------------------------------------------------------------
extern "C" void kernel_launch(void* const* d_in, const int* in_sizes, int n_in,
                              void* d_out, int out_size, void* d_ws, size_t ws_size,
                              hipStream_t stream) {
    const float* y    = (const float*)d_in[0];
    const float* wih0 = (const float*)d_in[1];
    const float* whh0 = (const float*)d_in[2];
    const float* b0   = (const float*)d_in[3];
    const float* bn0  = (const float*)d_in[4];
    const float* wih1 = (const float*)d_in[5];
    const float* whh1 = (const float*)d_in[6];
    const float* b1   = (const float*)d_in[7];
    const float* bn1  = (const float*)d_in[8];
    const float* lw   = (const float*)d_in[9];
    const float* lb   = (const float*)d_in[10];
    float* out = (float*)d_out;

    // workspace layout (256B aligned), total ~104 MB
    char* ws = (char*)d_ws;
    size_t off = 0;
    auto take = [&](size_t bytes) -> char* {
        char* p = ws + off;
        off = (off + bytes + 255) & ~(size_t)255;
        return p;
    };
    unsigned short* y_bf    = (unsigned short*)take((size_t)T_STEPS * H * 2);
    unsigned short* wih0_bf = (unsigned short*)take((size_t)H3 * H * 2);
    unsigned short* wih1_bf = (unsigned short*)take((size_t)H3 * H * 2);
    unsigned short* whh0T   = (unsigned short*)take((size_t)H3 * H * 2);
    unsigned short* whh1T   = (unsigned short*)take((size_t)H3 * H * 2);
    unsigned short* seq1_bf = (unsigned short*)take((size_t)T_STEPS * H * 2);
    float*          ig      = (float*)take((size_t)T_STEPS * H3 * 4);  // reused
    float*          hfin    = (float*)take((size_t)H * 4);

    // 1) conversions
    {
        int n = T_STEPS * H;
        cvt_f32_bf16<<<(n + 255) / 256, 256, 0, stream>>>(y, y_bf, n);
        int m = H3 * H;
        cvt_f32_bf16<<<(m + 255) / 256, 256, 0, stream>>>(wih0, wih0_bf, m);
        cvt_f32_bf16<<<(m + 255) / 256, 256, 0, stream>>>(wih1, wih1_bf, m);
        whh_transpose_bf16<<<(m + 255) / 256, 256, 0, stream>>>(whh0, whh0T);
        whh_transpose_bf16<<<(m + 255) / 256, 256, 0, stream>>>(whh1, whh1T);
    }

    dim3 gemm_grid(T_STEPS / 16, H3 / 64);   // 1024 x 18 wave-strips

    // 2) layer 0: ig0 = y @ w_ih_0^T + b_0 ; scan -> seq1 (bf16)
    wmma_gemm_bias<<<gemm_grid, 32, 0, stream>>>(y_bf, wih0_bf, b0, ig);
    gru_scan<<<1, H, 0, stream>>>(ig, whh0T, bn0, seq1_bf, hfin);

    // 3) layer 1: ig1 = seq1 @ w_ih_1^T + b_1 ; scan -> final h only
    wmma_gemm_bias<<<gemm_grid, 32, 0, stream>>>(seq1_bf, wih1_bf, b1, ig);
    gru_scan<<<1, H, 0, stream>>>(ig, whh1T, bn1, nullptr, hfin);

    // 4) head
    final_linear<<<1, N_EFF, 0, stream>>>(hfin, lw, lb, out);
}
